// PathPredictor_22771916604025
// MI455X (gfx1250) — matrix operand order repaired and checked by reading, not compiled, not run on hardware
//
#include <hip/hip_runtime.h>
#include <hip/hip_bf16.h>
#include <stdint.h>
#include <stddef.h>

#define N_NODES 8000
#define N_EDGES 256000
#define IN_DIM  32
#define HID     128
#define OUTC    (N_NODES + 1)   /* 8001 */
#define K1P     64              /* layer-1 K (34) zero-padded to 64 for 16x16x32 WMMA */

typedef __attribute__((ext_vector_type(16))) _Float16 v16h;
typedef __attribute__((ext_vector_type(8)))  _Float16 v8h;
typedef __attribute__((ext_vector_type(8)))  float    v8f;

/* ---------------- utility kernels ---------------- */

__global__ void k_zero4(float4* __restrict__ p, size_t n4) {
  size_t i = (size_t)blockIdx.x * blockDim.x + threadIdx.x;
  size_t stride = (size_t)gridDim.x * blockDim.x;
  float4 z = make_float4(0.f, 0.f, 0.f, 0.f);
  for (; i < n4; i += stride) p[i] = z;
}

__global__ void k_build_h0(const float* __restrict__ x, const int* __restrict__ cur,
                           const int* __restrict__ targ,
                           float* __restrict__ h32, _Float16* __restrict__ h16) {
  int c = *cur, t = *targ;
  size_t total = (size_t)N_NODES * K1P;
  size_t stride = (size_t)gridDim.x * blockDim.x;
  for (size_t idx = (size_t)blockIdx.x * blockDim.x + threadIdx.x; idx < total; idx += stride) {
    int i = (int)(idx >> 6);
    int k = (int)(idx & 63);
    float v = 0.f;
    if (k < IN_DIM)            v = x[(size_t)i * IN_DIM + k];
    else if (k == IN_DIM)      v = (i == c) ? 1.f : 0.f;
    else if (k == IN_DIM + 1)  v = (i == t) ? 1.f : 0.f;
    h32[idx] = v;
    h16[idx] = (_Float16)v;
  }
}

__global__ void k_deg(const int* __restrict__ dst, float* __restrict__ deg) {
  int e = blockIdx.x * blockDim.x + threadIdx.x;
  if (e < N_EDGES) atomicAdd(&deg[dst[e]], 1.0f);
}

__global__ void k_invdeg(const float* __restrict__ deg, float* __restrict__ invd) {
  int i = blockIdx.x * blockDim.x + threadIdx.x;
  if (i < N_NODES) invd[i] = 1.0f / fmaxf(deg[i], 1.0f);
}

/* scatter-sum over edges: agg[dst] += h[src]  (K = 1<<kshift floats per row) */
__global__ void k_scatter(const float* __restrict__ h, const int* __restrict__ src,
                          const int* __restrict__ dst, float* __restrict__ agg, int kshift) {
  size_t total  = ((size_t)N_EDGES) << kshift;
  size_t stride = (size_t)gridDim.x * blockDim.x;
  size_t kmask  = ((size_t)1 << kshift) - 1;
  for (size_t idx = (size_t)blockIdx.x * blockDim.x + threadIdx.x; idx < total; idx += stride) {
    int e = (int)(idx >> kshift);
    int k = (int)(idx & kmask);
    atomicAdd(&agg[(((size_t)dst[e]) << kshift) + k], h[(((size_t)src[e]) << kshift) + k]);
  }
}

/* agg16 = (agg32 * invdeg[row]) as f16 */
__global__ void k_scale_cvt(const float* __restrict__ agg32, const float* __restrict__ invd,
                            _Float16* __restrict__ agg16, int kshift) {
  size_t total  = ((size_t)N_NODES) << kshift;
  size_t stride = (size_t)gridDim.x * blockDim.x;
  for (size_t idx = (size_t)blockIdx.x * blockDim.x + threadIdx.x; idx < total; idx += stride)
    agg16[idx] = (_Float16)(agg32[idx] * invd[idx >> kshift]);
}

/* weights: src f32 [Kv][128] -> dst f16 TRANSPOSED [128][Kp], rows Kv..Kp zero.
 * Gives each WMMA lane two contiguous 8-half B pieces (2x ds/global b128). */
__global__ void k_cvtT(const float* __restrict__ s, _Float16* __restrict__ d, int Kp, int Kv) {
  int i = blockIdx.x * blockDim.x + threadIdx.x;   /* over 128*Kp */
  if (i < HID * Kp) {
    int n = i / Kp;
    int r = i - n * Kp;
    d[i] = (r < Kv) ? (_Float16)s[(size_t)r * HID + n] : (_Float16)0.f;
  }
}

/* fc_w [128][8001] -> wT [8001][128] */
__global__ void k_transpose_fcw(const float* __restrict__ w, float* __restrict__ wT) {
  size_t total  = (size_t)OUTC * HID;
  size_t stride = (size_t)gridDim.x * blockDim.x;
  for (size_t idx = (size_t)blockIdx.x * blockDim.x + threadIdx.x; idx < total; idx += stride) {
    int d = (int)(idx >> 7);
    int k = (int)(idx & 127);
    wT[idx] = w[(size_t)k * OUTC + d];
  }
}

/* ---------------- WMMA SAGE layer ----------------
 * out = relu(A0@B0 + A1@B1 + bias), M=8000, N=128.
 * A:(M,K) f16 in global; B supplied TRANSPOSED BT:(128,K) f16.
 * Both 16xK A-tiles (contiguous in memory) are staged into LDS with
 * global_load_async_to_lds_b128 (ASYNCcnt) once per block, then each of the
 * 8 waves computes one 16x16 N-tile with v_wmma_f32_16x16x32_f16. */
__global__ void __launch_bounds__(256)
k_sage_wmma(const _Float16* __restrict__ A0, const _Float16* __restrict__ BT0,
            const _Float16* __restrict__ A1, const _Float16* __restrict__ BT1,
            const float* __restrict__ bias, float* __restrict__ out32,
            _Float16* __restrict__ out16, int K) {
  __shared__ __align__(16) _Float16 As[2 * 16 * HID];   /* max 8 KB */
  const int tid = threadIdx.x;
  const int m0  = blockIdx.x * 16;

  /* async-copy the two contiguous 16xK f16 tiles (agg rows, h rows) into LDS */
  {
    unsigned lds_base = (unsigned)(uintptr_t)(void*)&As[0];
    int chunksPerPhase = (16 * K * 2) / 16;             /* 16-byte chunks */
    int nchunks = 2 * chunksPerPhase;
    unsigned long long g0 = (unsigned long long)(uintptr_t)(A0 + (size_t)m0 * K);
    unsigned long long g1 = (unsigned long long)(uintptr_t)(A1 + (size_t)m0 * K);
    for (int ch = tid; ch < nchunks; ch += 256) {
      int inPhase1 = ch >= chunksPerPhase;
      int c = inPhase1 ? (ch - chunksPerPhase) : ch;
      unsigned long long ga = (inPhase1 ? g1 : g0) + (unsigned long long)c * 16ull;
      unsigned ldso = lds_base + (unsigned)ch * 16u;
      asm volatile("global_load_async_to_lds_b128 %0, %1, off"
                   :: "v"(ldso), "v"(ga) : "memory");
    }
    asm volatile("s_wait_asynccnt 0x0" ::: "memory");
    __syncthreads();
  }

  const int wave = tid >> 5;
  const int lane = tid & 31;
  const int n0   = wave * 16;
  const int mrow = lane & 15;
  const int hi   = lane >> 4;      /* 0: lanes 0-15, 1: lanes 16-31 */
  const int ka   = hi * 8;         /* A K-offset inside each 16-group */
  const int kb   = hi * 16;        /* unused; B also uses ka */
  const int ncol = n0 + mrow;

  v8f c = {};
  for (int phase = 0; phase < 2; ++phase) {
    const _Float16* Arow = &As[phase * 16 * K + mrow * K];
    const _Float16* Brow = (phase ? BT1 : BT0) + (size_t)ncol * K;
    for (int k = 0; k < K; k += 32) {
      v8h alo = *(const v8h*)(Arow + k + ka);          /* ds_load_b128 */
      v8h ahi = *(const v8h*)(Arow + k + 16 + ka);
      v8h blo = *(const v8h*)(Brow + k + ka);          /* global_load_b128 */
      v8h bhi = *(const v8h*)(Brow + k + 16 + ka);
      v16h a = __builtin_shufflevector(alo, ahi, 0,1,2,3,4,5,6,7,8,9,10,11,12,13,14,15);
      v16h b = __builtin_shufflevector(blo, bhi, 0,1,2,3,4,5,6,7,8,9,10,11,12,13,14,15);
      c = __builtin_amdgcn_wmma_f32_16x16x32_f16(false, a, false, b, (short)0, c, false, false);
    }
  }

  float bv  = bias[ncol];
  int mbase = m0 + hi * 8;
#pragma unroll
  for (int r = 0; r < 8; ++r) {
    float v = c[r] + bv;
    v = v > 0.f ? v : 0.f;
    size_t o = (size_t)(mbase + r) * HID + ncol;
    out32[o] = v;
    out16[o] = (_Float16)v;
  }
}

/* ---------------- masked-softmax edge pipeline ---------------- */

__device__ __forceinline__ unsigned int fkey(float f) {
  unsigned int b = __float_as_uint(f);
  return (b & 0x80000000u) ? ~b : (b | 0x80000000u);
}
__device__ __forceinline__ float fdecode(unsigned int k) {
  unsigned int b = (k & 0x80000000u) ? (k & 0x7FFFFFFFu) : ~k;
  return __uint_as_float(b);
}

/* one wave32 per edge: l[e] = h3[src].wT[dst] + fc_b[dst]; rowmax[src] = max */
__global__ void k_edge_logit(const float* __restrict__ h3, const float* __restrict__ wT,
                             const float* __restrict__ fcb, const int* __restrict__ src,
                             const int* __restrict__ dst, float* __restrict__ elog,
                             unsigned int* __restrict__ rowmax) {
  int wv   = (int)(((size_t)blockIdx.x * blockDim.x + threadIdx.x) >> 5);
  int lane = threadIdx.x & 31;
  int nw   = (int)(((size_t)gridDim.x * blockDim.x) >> 5);
  for (int e = wv; e < N_EDGES; e += nw) {
    int s = src[e], d = dst[e];
    const float4* hp = (const float4*)(h3 + (size_t)s * HID);
    const float4* wp = (const float4*)(wT + (size_t)d * HID);
    float4 hv = hp[lane];
    float4 wk = wp[lane];
    float part = hv.x * wk.x + hv.y * wk.y + hv.z * wk.z + hv.w * wk.w;
    for (int o = 16; o > 0; o >>= 1) part += __shfl_xor(part, o, 32);
    if (lane == 0) {
      float l = part + fcb[d];
      elog[e] = l;
      atomicMax(&rowmax[s], fkey(l));
    }
  }
}

__global__ void k_edge_exp(const int* __restrict__ src, const float* __restrict__ elog,
                           const unsigned int* __restrict__ rowmax,
                           float* __restrict__ eexp, float* __restrict__ rowsum) {
  int e = blockIdx.x * blockDim.x + threadIdx.x;
  if (e < N_EDGES) {
    int s = src[e];
    float v = __expf(elog[e] - fdecode(rowmax[s]));
    eexp[e] = v;
    atomicAdd(&rowsum[s], v);
  }
}

__global__ void k_edge_write(const int* __restrict__ src, const int* __restrict__ dst,
                             const float* __restrict__ eexp, const float* __restrict__ rowsum,
                             float* __restrict__ out) {
  int e = blockIdx.x * blockDim.x + threadIdx.x;
  if (e < N_EDGES) {
    int s = src[e], d = dst[e];
    out[(size_t)s * OUTC + d] = eexp[e] / rowsum[s];
  }
}

/* ---------------- host launcher ---------------- */

extern "C" void kernel_launch(void* const* d_in, const int* in_sizes, int n_in,
                              void* d_out, int out_size, void* d_ws, size_t ws_size,
                              hipStream_t stream) {
  (void)in_sizes; (void)n_in; (void)out_size; (void)ws_size;

  const float* x      = (const float*)d_in[0];
  const int*   eidx   = (const int*)d_in[1];
  const int*   cur    = (const int*)d_in[2];
  const int*   targ   = (const int*)d_in[3];
  const float* w1_l   = (const float*)d_in[4];
  const float* w1_r   = (const float*)d_in[5];
  const float* b1     = (const float*)d_in[6];
  const float* w2_l   = (const float*)d_in[7];
  const float* w2_r   = (const float*)d_in[8];
  const float* b2     = (const float*)d_in[9];
  const float* w3_l   = (const float*)d_in[10];
  const float* w3_r   = (const float*)d_in[11];
  const float* b3     = (const float*)d_in[12];
  const float* fc_w   = (const float*)d_in[13];
  const float* fc_b   = (const float*)d_in[14];
  const int* src = eidx;
  const int* dst = eidx + N_EDGES;
  float* out = (float*)d_out;

  /* workspace layout (all chunks 16B-aligned) */
  char* w = (char*)d_ws;
  size_t o = 0;
  float*    h0_32 = (float*)(w + o);    o += (size_t)N_NODES * K1P * 4;
  _Float16* h0_16 = (_Float16*)(w + o); o += (size_t)N_NODES * K1P * 2;
  float*    agg32 = (float*)(w + o);    o += (size_t)N_NODES * HID * 4;
  _Float16* agg16 = (_Float16*)(w + o); o += (size_t)N_NODES * HID * 2;
  float*    h1_32 = (float*)(w + o);    o += (size_t)N_NODES * HID * 4;
  _Float16* h1_16 = (_Float16*)(w + o); o += (size_t)N_NODES * HID * 2;
  float*    h2_32 = (float*)(w + o);    o += (size_t)N_NODES * HID * 4;
  _Float16* h2_16 = (_Float16*)(w + o); o += (size_t)N_NODES * HID * 2;
  float*    h3_32 = (float*)(w + o);    o += (size_t)N_NODES * HID * 4;
  _Float16* h3_16 = (_Float16*)(w + o); o += (size_t)N_NODES * HID * 2;
  _Float16* w1lT  = (_Float16*)(w + o); o += (size_t)HID * K1P * 2;
  _Float16* w1rT  = (_Float16*)(w + o); o += (size_t)HID * K1P * 2;
  _Float16* w2lT  = (_Float16*)(w + o); o += (size_t)HID * HID * 2;
  _Float16* w2rT  = (_Float16*)(w + o); o += (size_t)HID * HID * 2;
  _Float16* w3lT  = (_Float16*)(w + o); o += (size_t)HID * HID * 2;
  _Float16* w3rT  = (_Float16*)(w + o); o += (size_t)HID * HID * 2;
  float*    fcwT  = (float*)(w + o);    o += (size_t)OUTC * HID * 4;
  o = (o + 15) & ~(size_t)15;
  float*        deg   = (float*)(w + o);        o += (size_t)N_NODES * 4;
  float*        invd  = (float*)(w + o);        o += (size_t)N_NODES * 4;
  unsigned int* rmax  = (unsigned int*)(w + o); o += (size_t)N_NODES * 4;
  float*        rsum  = (float*)(w + o);        o += (size_t)N_NODES * 4;
  float*        elog  = (float*)(w + o);        o += (size_t)N_EDGES * 4;
  float*        eexp  = (float*)(w + o);        o += (size_t)N_EDGES * 4;

  const int T = 256;
  /* 1. zero output (256 MB) + per-call accumulators */
  {
    size_t n4 = ((size_t)N_NODES * OUTC) / 4;
    k_zero4<<<16384, T, 0, stream>>>((float4*)out, n4);
    k_zero4<<<8, T, 0, stream>>>((float4*)deg,  N_NODES / 4);
    k_zero4<<<8, T, 0, stream>>>((float4*)rmax, N_NODES / 4);
    k_zero4<<<8, T, 0, stream>>>((float4*)rsum, N_NODES / 4);
  }
  /* 2. features, degrees, weight conversion (transposed f16) */
  k_build_h0<<<2048, T, 0, stream>>>(x, cur, targ, h0_32, h0_16);
  k_deg<<<(N_EDGES + T - 1) / T, T, 0, stream>>>(dst, deg);
  k_invdeg<<<(N_NODES + T - 1) / T, T, 0, stream>>>(deg, invd);
  k_cvtT<<<(HID * K1P + T - 1) / T, T, 0, stream>>>(w1_l, w1lT, K1P, IN_DIM + 2);
  k_cvtT<<<(HID * K1P + T - 1) / T, T, 0, stream>>>(w1_r, w1rT, K1P, IN_DIM + 2);
  k_cvtT<<<(HID * HID + T - 1) / T, T, 0, stream>>>(w2_l, w2lT, HID, HID);
  k_cvtT<<<(HID * HID + T - 1) / T, T, 0, stream>>>(w2_r, w2rT, HID, HID);
  k_cvtT<<<(HID * HID + T - 1) / T, T, 0, stream>>>(w3_l, w3lT, HID, HID);
  k_cvtT<<<(HID * HID + T - 1) / T, T, 0, stream>>>(w3_r, w3rT, HID, HID);
  k_transpose_fcw<<<2048, T, 0, stream>>>(fc_w, fcwT);

  /* 3. SAGE layer 1 (K=64 padded) */
  k_zero4<<<2048, T, 0, stream>>>((float4*)agg32, (size_t)N_NODES * K1P / 4);
  k_scatter<<<4096, T, 0, stream>>>(h0_32, src, dst, agg32, 6);
  k_scale_cvt<<<2048, T, 0, stream>>>(agg32, invd, agg16, 6);
  k_sage_wmma<<<N_NODES / 16, T, 0, stream>>>(agg16, w1lT, h0_16, w1rT, b1, h1_32, h1_16, K1P);

  /* 4. SAGE layer 2 (K=128) */
  k_zero4<<<2048, T, 0, stream>>>((float4*)agg32, (size_t)N_NODES * HID / 4);
  k_scatter<<<4096, T, 0, stream>>>(h1_32, src, dst, agg32, 7);
  k_scale_cvt<<<2048, T, 0, stream>>>(agg32, invd, agg16, 7);
  k_sage_wmma<<<N_NODES / 16, T, 0, stream>>>(agg16, w2lT, h1_16, w2rT, b2, h2_32, h2_16, HID);

  /* 5. SAGE layer 3 (K=128) */
  k_zero4<<<2048, T, 0, stream>>>((float4*)agg32, (size_t)N_NODES * HID / 4);
  k_scatter<<<4096, T, 0, stream>>>(h2_32, src, dst, agg32, 7);
  k_scale_cvt<<<2048, T, 0, stream>>>(agg32, invd, agg16, 7);
  k_sage_wmma<<<N_NODES / 16, T, 0, stream>>>(agg16, w3lT, h2_16, w3rT, b3, h3_32, h3_16, HID);

  /* 6. edge-only masked softmax (dense GEMM+softmax algebraically eliminated) */
  k_edge_logit<<<N_EDGES / 8, T, 0, stream>>>(h3_32, fcwT, fc_b, src, dst, elog, rmax);
  k_edge_exp<<<(N_EDGES + T - 1) / T, T, 0, stream>>>(src, elog, rmax, eexp, rsum);
  k_edge_write<<<(N_EDGES + T - 1) / T, T, 0, stream>>>(src, dst, eexp, rsum, out);
}
